// CrystalTransformerEncoder_11063835755127
// MI455X (gfx1250) — matrix operand order repaired and checked by reading, not compiled, not run on hardware
//
#include <hip/hip_runtime.h>
#include <math.h>

#define NN 8000
#define EE 200000
#define BBATCH 16
#define DD 256
#define HH 4
#define HD 64
#define LL 4
#define NFRE 100

typedef __attribute__((ext_vector_type(16))) __bf16 v16bf;
typedef __attribute__((ext_vector_type(8)))  __bf16 v8bf;
typedef __attribute__((ext_vector_type(8)))  float  v8f;

// ---------------- helpers ----------------
__device__ __forceinline__ unsigned fkey(float f) {
    unsigned u = __float_as_uint(f);
    return (u & 0x80000000u) ? ~u : (u | 0x80000000u);
}
__device__ __forceinline__ float funkey(unsigned u) {
    unsigned v = (u & 0x80000000u) ? (u & 0x7fffffffu) : ~u;
    return __uint_as_float(v);
}
__device__ __forceinline__ float gelu_exact(float x) {
    return 0.5f * x * (1.0f + erff(x * 0.70710678118654752f));
}

// ---------------- generic bf16 WMMA GEMM ----------------
// C[M][N] = A[M][K](bf16,row-major) @ Wt[N][K](bf16) + bias[N]; optional GELU.
// Block: 256 threads = 8 waves. Wave (msub,ngrp): 16 rows x 64 cols.
// Block tile: 32 rows x 256 cols.  Requires M%32==0, N%256==0, K%32==0.
__global__ void gemm_bf16_kernel(const __bf16* __restrict__ A,
                                 const __bf16* __restrict__ Wt,
                                 const float* __restrict__ bias,
                                 __bf16* __restrict__ Cbf,
                                 float* __restrict__ Cf,
                                 int M, int K, int N, int act)
{
    const int tid   = threadIdx.x;
    const int wave  = tid >> 5;
    const int lane  = tid & 31;
    const int msub  = wave & 1;
    const int ngrp  = wave >> 1;
    const int lhalf = lane >> 4;
    const int l15   = lane & 15;

    const int m0 = blockIdx.x * 32 + msub * 16;
    const int n0 = blockIdx.y * 256 + ngrp * 64;

    const int klo   = lhalf ? 16 : 0;
    const int khi   = lhalf ? 24 : 8;
    const int koffB = lhalf ? 16 : 0;

    const __bf16* Ap  = A  + (size_t)(m0 + l15) * K;
    const __bf16* Wp0 = Wt + (size_t)(n0 +  0 + l15) * K + koffB;
    const __bf16* Wp1 = Wt + (size_t)(n0 + 16 + l15) * K + koffB;
    const __bf16* Wp2 = Wt + (size_t)(n0 + 32 + l15) * K + koffB;
    const __bf16* Wp3 = Wt + (size_t)(n0 + 48 + l15) * K + koffB;

    v8f acc0 = {}, acc1 = {}, acc2 = {}, acc3 = {};

    for (int k0 = 0; k0 < K; k0 += 32) {
        v8bf alo = *(const v8bf*)(Ap + k0 + klo);
        v8bf ahi = *(const v8bf*)(Ap + k0 + khi);
        v16bf a = __builtin_shufflevector(alo, ahi,
                    0,1,2,3,4,5,6,7,8,9,10,11,12,13,14,15);
        v16bf b0 = *(const v16bf*)(Wp0 + k0);
        v16bf b1 = *(const v16bf*)(Wp1 + k0);
        v16bf b2 = *(const v16bf*)(Wp2 + k0);
        v16bf b3 = *(const v16bf*)(Wp3 + k0);
        acc0 = __builtin_amdgcn_wmma_f32_16x16x32_bf16(false, a, false, b0, (short)0, acc0, false, false);
        acc1 = __builtin_amdgcn_wmma_f32_16x16x32_bf16(false, a, false, b1, (short)0, acc1, false, false);
        acc2 = __builtin_amdgcn_wmma_f32_16x16x32_bf16(false, a, false, b2, (short)0, acc2, false, false);
        acc3 = __builtin_amdgcn_wmma_f32_16x16x32_bf16(false, a, false, b3, (short)0, acc3, false, false);
    }

    const int mrow0 = m0 + lhalf * 8;
    auto epi = [&](v8f acc, int t) {
        const int col = n0 + t * 16 + l15;
        const float bs = bias ? bias[col] : 0.0f;
        #pragma unroll
        for (int j = 0; j < 8; ++j) {
            float v = acc[j] + bs;
            if (act == 1) v = gelu_exact(v);
            size_t o = (size_t)(mrow0 + j) * N + col;
            if (Cf)  Cf[o]  = v;
            if (Cbf) Cbf[o] = (__bf16)v;
        }
    };
    epi(acc0, 0); epi(acc1, 1); epi(acc2, 2); epi(acc3, 3);
}

// ---------------- weight convert + transpose (f32[K][N] -> bf16[N][Kpad]) ----------------
__global__ void k_wt(const float* __restrict__ W, __bf16* __restrict__ Wt,
                     int Ksrc, int Kp, int Nc)
{
    long long idx = (long long)blockIdx.x * blockDim.x + threadIdx.x;
    if (idx >= (long long)Nc * Kp) return;
    int n = (int)(idx / Kp);
    int k = (int)(idx % Kp);
    Wt[idx] = (k < Ksrc) ? (__bf16)W[(size_t)k * Nc + n] : (__bf16)0.0f;
}

// ---------------- lattice features: l_feat[b][p] = (L^T L)[iu] ----------------
__global__ void k_lattice(const float* __restrict__ lat, float* __restrict__ lfeat)
{
    int t = threadIdx.x;
    if (t >= BBATCH * 6) return;
    const int iu0[6] = {0,0,0,1,1,2};
    const int iu1[6] = {0,1,2,1,2,2};
    int b = t / 6, p = t % 6;
    int j = iu0[p], k = iu1[p];
    float s = 0.0f;
    for (int i = 0; i < 3; ++i)
        s += lat[b*9 + i*3 + j] * lat[b*9 + i*3 + k];
    lfeat[b*6 + p] = s;
}

// ---------------- node embed: h_n = [emb, lfeat] @ node_W + node_b ----------------
__global__ void k_hn_init(const int* __restrict__ types, const float* __restrict__ emb,
                          const float* __restrict__ lfeat, const int* __restrict__ batch,
                          const float* __restrict__ nW, const float* __restrict__ nb,
                          float* __restrict__ hn, __bf16* __restrict__ hnb)
{
    long long idx = (long long)blockIdx.x * blockDim.x + threadIdx.x;
    if (idx >= (long long)NN * DD) return;
    int n = (int)(idx >> 8), d = (int)(idx & 255);
    const float* e = emb + (size_t)types[n] * DD;
    float v = nb[d];
    for (int k = 0; k < DD; ++k) v += e[k] * nW[(size_t)k * DD + d];
    const float* lf = lfeat + batch[n] * 6;
    for (int j = 0; j < 6; ++j) v += lf[j] * nW[(size_t)(DD + j) * DD + d];
    hn[idx] = v;
    hnb[idx] = (__bf16)v;
}

// ---------------- edge displacement ----------------
__global__ void k_disp(const float* __restrict__ frac, const int* __restrict__ src,
                       const int* __restrict__ dst, const float* __restrict__ shift,
                       float* __restrict__ disp)
{
    long long idx = (long long)blockIdx.x * blockDim.x + threadIdx.x;
    if (idx >= (long long)EE * 3) return;
    int e = (int)(idx / 3), c = (int)(idx % 3);
    float d = frac[(size_t)dst[e]*3 + c] - frac[(size_t)src[e]*3 + c] + shift[idx];
    d -= floorf(d);
    disp[idx] = d;
}

// ---------------- sinusoidal edge features (bf16, padded to 608 cols) ----------------
__global__ void k_edge_feat(const float* __restrict__ disp, __bf16* __restrict__ ef)
{
    long long idx = (long long)blockIdx.x * blockDim.x + threadIdx.x;
    if (idx >= (long long)EE * 608) return;
    int e = (int)(idx / 608), col = (int)(idx % 608);
    float v = 0.0f;
    if (col < 600) {
        bool iscos = col >= 300;
        int cc = iscos ? col - 300 : col;
        int c = cc / NFRE, f = cc % NFRE;
        float ang = disp[(size_t)e*3 + c] * (6.28318530717958647692f * (float)f);
        v = iscos ? cosf(ang) : sinf(ang);
    }
    ef[idx] = (__bf16)v;
}

// ---------------- gather/concat into GEMM A buffer ----------------
// full==1: Aq[e] = [hn[nodes[e]] (256), he[e] (256)]  (E*64 uint4 chunks)
// full==0: overwrite first 256 cols only                 (E*32 uint4 chunks)
__global__ void k_gather(const __bf16* __restrict__ hnb, const __bf16* __restrict__ he,
                         const int* __restrict__ nodes, __bf16* __restrict__ Aq, int full)
{
    long long idx = (long long)blockIdx.x * blockDim.x + threadIdx.x;
    if (full) {
        if (idx >= (long long)EE * 64) return;
        long long e = idx >> 6; int c = (int)(idx & 63);
        uint4 v;
        if (c < 32) v = ((const uint4*)(hnb + (size_t)nodes[e] * DD))[c];
        else        v = ((const uint4*)(he  + (size_t)e * DD))[c - 32];
        ((uint4*)(Aq + (size_t)e * 512))[c] = v;
    } else {
        if (idx >= (long long)EE * 32) return;
        long long e = idx >> 5; int c = (int)(idx & 31);
        ((uint4*)(Aq + (size_t)e * 512))[c] =
            ((const uint4*)(hnb + (size_t)nodes[e] * DD))[c];
    }
}

// ---------------- zero segment-softmax state ----------------
__global__ void k_zero_seg(float* __restrict__ attn, unsigned* __restrict__ m_u,
                           float* __restrict__ den)
{
    long long idx = (long long)blockIdx.x * blockDim.x + threadIdx.x;
    if (idx < (long long)NN * DD) attn[idx] = 0.0f;
    if (idx < (long long)NN * HH) { m_u[idx] = 0u; den[idx] = 0.0f; }
}

// ---------------- scores + segment max ----------------
__global__ void k_scores(const __bf16* __restrict__ Q, const __bf16* __restrict__ KV,
                         const int* __restrict__ src, float* __restrict__ scores,
                         unsigned* __restrict__ m_u)
{
    long long idx = (long long)blockIdx.x * blockDim.x + threadIdx.x;
    if (idx >= (long long)EE * HH) return;
    long long e = idx >> 2; int h = (int)(idx & 3);
    const v8bf* qv = (const v8bf*)(Q  + (size_t)e * DD  + h * HD);
    const v8bf* kv = (const v8bf*)(KV + (size_t)e * 512 + h * HD);
    float s = 0.0f;
    #pragma unroll
    for (int c = 0; c < 8; ++c) {
        v8bf qa = qv[c], ka = kv[c];
        #pragma unroll
        for (int i = 0; i < 8; ++i) s += (float)qa[i] * (float)ka[i];
    }
    s *= 0.125f; // 1/sqrt(64)
    scores[idx] = s;
    atomicMax(m_u + (size_t)src[e] * HH + h, fkey(s));
}

// ---------------- exp + segment sum ----------------
__global__ void k_expsum(const int* __restrict__ src, const unsigned* __restrict__ m_u,
                         float* __restrict__ scores, float* __restrict__ den)
{
    long long idx = (long long)blockIdx.x * blockDim.x + threadIdx.x;
    if (idx >= (long long)EE * HH) return;
    long long e = idx >> 2; int h = (int)(idx & 3);
    float mv = funkey(m_u[(size_t)src[e] * HH + h]);
    float ex = expf(scores[idx] - mv);
    scores[idx] = ex;
    atomicAdd(den + (size_t)src[e] * HH + h, ex);
}

// ---------------- weighted V scatter-add ----------------
__global__ void k_attn_scatter(const int* __restrict__ src, const float* __restrict__ scores,
                               const float* __restrict__ den, const __bf16* __restrict__ KV,
                               float* __restrict__ attn)
{
    long long idx = (long long)blockIdx.x * blockDim.x + threadIdx.x;
    if (idx >= (long long)EE * DD) return;
    long long e = idx >> 8; int d = (int)(idx & 255);
    int h = d >> 6;
    int s = src[e];
    float w = scores[e * HH + h] / den[(size_t)s * HH + h];
    float v = (float)KV[(size_t)e * 512 + 256 + d];
    atomicAdd(attn + (size_t)s * DD + d, w * v);
}

// ---------------- f32 -> bf16 ----------------
__global__ void k_cvt_bf16(const float* __restrict__ in, __bf16* __restrict__ out, long long n)
{
    long long idx = (long long)blockIdx.x * blockDim.x + threadIdx.x;
    if (idx < n) out[idx] = (__bf16)in[idx];
}

// ---------------- residual + layernorm (one block per row, D=256) ----------------
__global__ void k_ln(float* __restrict__ hn, const float* __restrict__ resid,
                     const float* __restrict__ g, const float* __restrict__ b,
                     __bf16* __restrict__ hnb)
{
    __shared__ float red[DD];
    int row = blockIdx.x, tid = threadIdx.x;
    size_t o = (size_t)row * DD + tid;
    float x = hn[o] + resid[o];
    red[tid] = x; __syncthreads();
    for (int s = 128; s > 0; s >>= 1) { if (tid < s) red[tid] += red[tid + s]; __syncthreads(); }
    float mu = red[0] * (1.0f / DD);
    __syncthreads();
    float d = x - mu; red[tid] = d * d; __syncthreads();
    for (int s = 128; s > 0; s >>= 1) { if (tid < s) red[tid] += red[tid + s]; __syncthreads(); }
    float var = red[0] * (1.0f / DD);
    float y = d * rsqrtf(var + 1e-5f) * g[tid] + b[tid];
    hn[o] = y;
    hnb[o] = (__bf16)y;
}

// ---------------- host ----------------
static inline void launch_gemm(hipStream_t s, const __bf16* A, const __bf16* Wt,
                               const float* bias, __bf16* Cbf, float* Cf,
                               int M, int K, int N, int act)
{
    dim3 grid(M / 32, N / 256);
    gemm_bf16_kernel<<<grid, dim3(256), 0, s>>>(A, Wt, bias, Cbf, Cf, M, K, N, act);
}

#define GRID1(n) dim3((unsigned)(((long long)(n) + 255) / 256)), dim3(256)

extern "C" void kernel_launch(void* const* d_in, const int* in_sizes, int n_in,
                              void* d_out, int out_size, void* d_ws, size_t ws_size,
                              hipStream_t stream)
{
    const int*   atom_types = (const int*)d_in[0];
    const float* lattices   = (const float*)d_in[1];
    const float* frac       = (const float*)d_in[2];
    const int*   edge_index = (const int*)d_in[3];
    const float* shifts     = (const float*)d_in[4];
    const int*   batch_ids  = (const int*)d_in[5];
    const float* atom_emb   = (const float*)d_in[6];
    const float* node_W     = (const float*)d_in[7];
    const float* node_b     = (const float*)d_in[8];
    const float* edge_W     = (const float*)d_in[9];
    const float* edge_b     = (const float*)d_in[10];
    const float* Wq   = (const float*)d_in[11];
    const float* bq   = (const float*)d_in[12];
    const float* Wkv  = (const float*)d_in[13];
    const float* bkv  = (const float*)d_in[14];
    const float* Wo   = (const float*)d_in[15];
    const float* bo   = (const float*)d_in[16];
    const float* ln1g = (const float*)d_in[17];
    const float* ln1b = (const float*)d_in[18];
    const float* W1   = (const float*)d_in[19];
    const float* b1   = (const float*)d_in[20];
    const float* W2   = (const float*)d_in[21];
    const float* b2   = (const float*)d_in[22];
    const float* ln2g = (const float*)d_in[23];
    const float* ln2b = (const float*)d_in[24];

    const int* src = edge_index;
    const int* dst = edge_index + EE;

    // workspace carve-up
    char* ws = (char*)d_ws;
    size_t off = 0;
    auto alloc = [&](size_t bytes) -> char* {
        char* p = ws + off; off += (bytes + 255) & ~(size_t)255; return p;
    };
    float*  hn_f32  = (float*) alloc((size_t)NN * DD * 4);
    __bf16* hn_bf   = (__bf16*)alloc((size_t)NN * DD * 2);
    __bf16* he_bf   = (__bf16*)alloc((size_t)EE * DD * 2);
    __bf16* Aq      = (__bf16*)alloc((size_t)EE * 608 * 2); // also edge_feat buffer
    __bf16* Qbf     = (__bf16*)alloc((size_t)EE * DD * 2);
    __bf16* KVbf    = (__bf16*)alloc((size_t)EE * 512 * 2);
    float*  scores  = (float*) alloc((size_t)EE * HH * 4);
    unsigned* m_u   = (unsigned*)alloc((size_t)NN * HH * 4);
    float*  den     = (float*) alloc((size_t)NN * HH * 4);
    float*  attn    = (float*) alloc((size_t)NN * DD * 4);
    __bf16* attn_bf = (__bf16*)alloc((size_t)NN * DD * 2);
    float*  tmp_f32 = (float*) alloc((size_t)NN * DD * 4);
    __bf16* ffn1_bf = (__bf16*)alloc((size_t)NN * 1024 * 2);
    float*  disp    = (float*) alloc((size_t)EE * 3 * 4);
    float*  lfeat   = (float*) alloc((size_t)BBATCH * 6 * 4);
    __bf16* WeT  = (__bf16*)alloc((size_t)DD * 608 * 2);
    __bf16* WqT  = (__bf16*)alloc((size_t)LL * DD * 512 * 2);
    __bf16* WkvT = (__bf16*)alloc((size_t)LL * 512 * 512 * 2);
    __bf16* WoT  = (__bf16*)alloc((size_t)LL * DD * DD * 2);
    __bf16* W1T  = (__bf16*)alloc((size_t)LL * 1024 * DD * 2);
    __bf16* W2T  = (__bf16*)alloc((size_t)LL * DD * 1024 * 2);

    // weight conversion (transposed bf16)
    k_wt<<<GRID1((long long)DD * 608), 0, stream>>>(edge_W, WeT, 600, 608, DD);
    for (int l = 0; l < LL; ++l) {
        k_wt<<<GRID1((long long)DD * 512), 0, stream>>>(Wq + (size_t)l*512*DD,  WqT + (size_t)l*DD*512,  512, 512, DD);
        k_wt<<<GRID1((long long)512 * 512), 0, stream>>>(Wkv + (size_t)l*512*512, WkvT + (size_t)l*512*512, 512, 512, 512);
        k_wt<<<GRID1((long long)DD * DD), 0, stream>>>(Wo + (size_t)l*DD*DD,   WoT + (size_t)l*DD*DD,   DD, DD, DD);
        k_wt<<<GRID1((long long)1024 * DD), 0, stream>>>(W1 + (size_t)l*DD*1024, W1T + (size_t)l*1024*DD, DD, DD, 1024);
        k_wt<<<GRID1((long long)DD * 1024), 0, stream>>>(W2 + (size_t)l*1024*DD, W2T + (size_t)l*DD*1024, 1024, 1024, DD);
    }

    // node / edge embeddings
    k_lattice<<<dim3(1), dim3(128), 0, stream>>>(lattices, lfeat);
    k_hn_init<<<GRID1((long long)NN * DD), 0, stream>>>(atom_types, atom_emb, lfeat, batch_ids,
                                                        node_W, node_b, hn_f32, hn_bf);
    k_disp<<<GRID1((long long)EE * 3), 0, stream>>>(frac, src, dst, shifts, disp);
    k_edge_feat<<<GRID1((long long)EE * 608), 0, stream>>>(disp, Aq);
    launch_gemm(stream, Aq, WeT, edge_b, he_bf, nullptr, EE, 608, DD, 0);

    for (int l = 0; l < LL; ++l) {
        // Q = [h_n[src], h_e] @ Wq + bq
        k_gather<<<GRID1((long long)EE * 64), 0, stream>>>(hn_bf, he_bf, src, Aq, 1);
        launch_gemm(stream, Aq, WqT + (size_t)l*DD*512, bq + (size_t)l*DD, Qbf, nullptr,
                    EE, 512, DD, 0);
        // KV = [h_n[dst], h_e] @ Wkv + bkv
        k_gather<<<GRID1((long long)EE * 32), 0, stream>>>(hn_bf, he_bf, dst, Aq, 0);
        launch_gemm(stream, Aq, WkvT + (size_t)l*512*512, bkv + (size_t)l*512, KVbf, nullptr,
                    EE, 512, 512, 0);
        // segment softmax over src
        k_zero_seg<<<GRID1((long long)NN * DD), 0, stream>>>(attn, m_u, den);
        k_scores<<<GRID1((long long)EE * HH), 0, stream>>>(Qbf, KVbf, src, scores, m_u);
        k_expsum<<<GRID1((long long)EE * HH), 0, stream>>>(src, m_u, scores, den);
        k_attn_scatter<<<GRID1((long long)EE * DD), 0, stream>>>(src, scores, den, KVbf, attn);
        // out @ Wo + bo ; residual+LN
        k_cvt_bf16<<<GRID1((long long)NN * DD), 0, stream>>>(attn, attn_bf, (long long)NN * DD);
        launch_gemm(stream, attn_bf, WoT + (size_t)l*DD*DD, bo + (size_t)l*DD, nullptr, tmp_f32,
                    NN, DD, DD, 0);
        k_ln<<<dim3(NN), dim3(DD), 0, stream>>>(hn_f32, tmp_f32, ln1g + (size_t)l*DD,
                                                ln1b + (size_t)l*DD, hn_bf);
        // FFN: gelu(h @ W1 + b1) @ W2 + b2 ; residual+LN
        launch_gemm(stream, hn_bf, W1T + (size_t)l*1024*DD, b1 + (size_t)l*1024, ffn1_bf, nullptr,
                    NN, DD, 1024, 1);
        launch_gemm(stream, ffn1_bf, W2T + (size_t)l*DD*1024, b2 + (size_t)l*DD, nullptr, tmp_f32,
                    NN, 1024, DD, 0);
        k_ln<<<dim3(NN), dim3(DD), 0, stream>>>(hn_f32, tmp_f32, ln2g + (size_t)l*DD,
                                                ln2b + (size_t)l*DD, hn_bf);
    }

    hipMemcpyAsync(d_out, hn_f32, (size_t)NN * DD * sizeof(float),
                   hipMemcpyDeviceToDevice, stream);
}
